// VRNN_5866925326838
// MI455X (gfx1250) — matrix operand order repaired and checked by reading, not compiled, not run on hardware
//
#include <hip/hip_runtime.h>

// ---- problem dims ----
#define XD_ 256
#define HD_ 1024
#define ZD_ 256
#define RD_ 2048
#define BB_ 64
#define TT_ 128

typedef __bf16 v16bf __attribute__((ext_vector_type(16)));
typedef __bf16 v8bf  __attribute__((ext_vector_type(8)));
typedef float  v8f   __attribute__((ext_vector_type(8)));

// =====================================================================
// WMMA GEMM:  C(M,N) = act(A(M,K) @ W(K,N) + bias)
// W is supplied transposed (WT: N x K, row-major, bf16) so both A and B
// fragments are contiguous loads using the same contiguous k order for A and B:
//   A (16x32): lanes 0-15 -> M=lane, K = k..k+15 contiguous
//              lanes 16-31 -> M=lane-16, K = k+16..k+31 contiguous
//   B (32x16): lanes 0-15 -> N=lane, K = k..k+15 contiguous
//              lanes 16-31 -> N=lane-16, K = k+16..k+31 contiguous
// Each wave computes a 32(M) x 64(N) strip: 2 A fragments x 4 B tiles
// -> 8 back-to-back v_wmma per K-step, 12 b128 loads (load:wmma 1.5:1).
// Block = 256 threads = 8 waves arranged 2(M) x 4(N) -> 64 x 256 tile.
// ACT: 0=identity 1=relu.  OUT: 0=f32, 1=bf16, 2=f32 with (t,b)->(b,t)
// row permutation (input rows ordered t*Bdim+b).
// =====================================================================
template<int ACT, int OUT>
__global__ __launch_bounds__(256) void gemm_wmma(
    const __bf16* __restrict__ A, const __bf16* __restrict__ WT,
    const float* __restrict__ bias, void* __restrict__ outp,
    int M, int K, int N, int Bdim, int Tdim)
{
    const int lane = threadIdx.x & 31;
    const int wave = threadIdx.x >> 5;
    const int row0 = blockIdx.y * 64  + (wave >> 2) * 32;   // 2 waves over M
    const int col0 = blockIdx.x * 256 + (wave & 3) * 64;    // 4 waves over N
    const int hi   = lane >> 4;     // lane half (0/1)
    const int l16  = lane & 15;

    const __bf16* __restrict__ arow0 = A  + (size_t)(row0 + l16) * K;
    const __bf16* __restrict__ arow1 = arow0 + (size_t)16 * K;
    const __bf16* __restrict__ brow0 = WT + (size_t)(col0 + l16) * K;
    const size_t bstep = (size_t)16 * K;    // next N-tile = +16 rows of WT

    v8f acc0[4] = {};
    v8f acc1[4] = {};
    for (int k = 0; k < K; k += 32) {
        const int ka = k + hi * 16;
        __builtin_prefetch(arow0 + ka + 256, 0, 1);           // global_prefetch_b8
        __builtin_prefetch(arow1 + ka + 256, 0, 1);
        v8bf a0lo = *(const v8bf*)(arow0 + ka);               // K = ka..ka+7
        v8bf a0hi = *(const v8bf*)(arow0 + ka + 8);           // K = ka+8..ka+15
        v8bf a1lo = *(const v8bf*)(arow1 + ka);
        v8bf a1hi = *(const v8bf*)(arow1 + ka + 8);
        v16bf a0, a1;
#pragma unroll
        for (int i = 0; i < 8; ++i) {
            a0[i] = a0lo[i]; a0[i + 8] = a0hi[i];
            a1[i] = a1lo[i]; a1[i + 8] = a1hi[i];
        }
#pragma unroll
        for (int j = 0; j < 4; ++j) {
            const __bf16* bp = brow0 + (size_t)j * bstep + k + hi * 16;
            __builtin_prefetch(bp + 256, 0, 1);
            v16bf b = *(const v16bf*)bp;
            acc0[j] = __builtin_amdgcn_wmma_f32_16x16x32_bf16(
                          false, a0, false, b, (short)0, acc0[j], false, false);
            acc1[j] = __builtin_amdgcn_wmma_f32_16x16x32_bf16(
                          false, a1, false, b, (short)0, acc1[j], false, false);
        }
    }

#pragma unroll
    for (int j = 0; j < 4; ++j) {
        const int n  = col0 + j * 16 + l16;    // C: col = lane%16
        const float bv = bias[n];
#pragma unroll
        for (int r = 0; r < 8; ++r) {
            // C: VGPR r -> M = r + 8*(lane/16); second tile at +16 rows
#pragma unroll
            for (int mtile = 0; mtile < 2; ++mtile) {
                int row = row0 + mtile * 16 + hi * 8 + r;
                float v = (mtile ? acc1[j][r] : acc0[j][r]) + bv;
                if (ACT == 1) v = fmaxf(v, 0.0f);
                if (OUT == 0)      ((float*)outp)[(size_t)row * N + n] = v;
                else if (OUT == 1) ((__bf16*)outp)[(size_t)row * N + n] = (__bf16)v;
                else {
                    int bi = row % Bdim, ti = row / Bdim;
                    ((float*)outp)[((size_t)bi * Tdim + ti) * N + n] = v;
                }
            }
        }
    }
}

// ---- weight prep: f32 (K x N) -> bf16 transposed (N x K), LDS tiled ----
__global__ __launch_bounds__(256) void transpose_cast_bf16(
    __bf16* __restrict__ dst, const float* __restrict__ src, int K, int N)
{
    __shared__ float tile[32][33];
    const int k0 = blockIdx.x * 32, n0 = blockIdx.y * 32;
    const int tx = threadIdx.x & 31, ty = threadIdx.x >> 5;   // 32x8
    for (int i = ty; i < 32; i += 8) {
        int k = k0 + i, n = n0 + tx;
        tile[i][tx] = (k < K && n < N) ? src[(size_t)k * N + n] : 0.0f;
    }
    __syncthreads();
    for (int i = ty; i < 32; i += 8) {
        int n = n0 + i, k = k0 + tx;
        if (n < N && k < K) dst[(size_t)n * K + k] = (__bf16)tile[tx][i];
    }
}

__global__ void cast_f32_bf16(__bf16* __restrict__ dst,
                              const float* __restrict__ src, int n)
{
    int i = blockIdx.x * blockDim.x + threadIdx.x;
    if (i < n) dst[i] = (__bf16)src[i];
}

__global__ void zero_f32(float* p, int n)
{ int i = blockIdx.x * blockDim.x + threadIdx.x; if (i < n) p[i] = 0.0f; }

__global__ void zero_bf16(__bf16* p, int n)
{ int i = blockIdx.x * blockDim.x + threadIdx.x; if (i < n) p[i] = (__bf16)0.0f; }

// strided bf16 sub-matrix copy (for concats)
__global__ void copy_bf16_blk(__bf16* __restrict__ dst, int dld,
                              const __bf16* __restrict__ src, int sld,
                              int rows, int cols)
{
    int i = blockIdx.x * blockDim.x + threadIdx.x;
    if (i >= rows * cols) return;
    int r = i / cols, c = i % cols;
    dst[(size_t)r * dld + c] = src[(size_t)r * sld + c];
}

__device__ __forceinline__ float softplus_(float x)
{ return (x > 20.0f) ? x : log1pf(__expf(x)); }
__device__ __forceinline__ float sigmoid_(float x)
{ return 1.0f / (1.0f + __expf(-x)); }

// o (64 x 2*ZD f32) -> mu, sd at stats slot t
__global__ void head_prior(const float* __restrict__ o,
                           float* __restrict__ mu, float* __restrict__ sd)
{
    int i = blockIdx.x * blockDim.x + threadIdx.x;   // 64*256
    if (i >= BB_ * ZD_) return;
    int r = i / ZD_, c = i % ZD_;
    mu[i] = o[(size_t)r * (2 * ZD_) + c];
    sd[i] = softplus_(o[(size_t)r * (2 * ZD_) + ZD_ + c]) + 1e-6f;
}

// posterior head: also z = mu + sd*eps_t  (z stored bf16 for phi_z MLP)
__global__ void head_post(const float* __restrict__ o,
                          const float* __restrict__ eps_t, int eps_ld,
                          float* __restrict__ mu, float* __restrict__ sd,
                          __bf16* __restrict__ z)
{
    int i = blockIdx.x * blockDim.x + threadIdx.x;
    if (i >= BB_ * ZD_) return;
    int r = i / ZD_, c = i % ZD_;
    float m = o[(size_t)r * (2 * ZD_) + c];
    float s = softplus_(o[(size_t)r * (2 * ZD_) + ZD_ + c]) + 1e-6f;
    mu[i] = m; sd[i] = s;
    z[i] = (__bf16)(m + s * eps_t[(size_t)r * eps_ld + c]);
}

// GRU gate fusion: h <- (1-z)*n + z*h ; also refresh bf16 copy of h
__global__ void gru_combine(const float* __restrict__ gi,
                            const float* __restrict__ gh,
                            float* __restrict__ h, __bf16* __restrict__ hb)
{
    int i = blockIdx.x * blockDim.x + threadIdx.x;   // 64*2048
    if (i >= BB_ * RD_) return;
    int r = i / RD_, c = i % RD_;
    size_t base = (size_t)r * (3 * RD_) + c;
    float ir = gi[base], iz = gi[base + RD_], in_ = gi[base + 2 * RD_];
    float hr = gh[base], hz = gh[base + RD_], hn  = gh[base + 2 * RD_];
    float rr = sigmoid_(ir + hr);
    float zt = sigmoid_(iz + hz);
    float nn = tanhf(in_ + rr * hn);
    float hv = (1.0f - zt) * nn + zt * h[i];
    h[i] = hv; hb[i] = (__bf16)hv;
}

// KLD over (T,B,ZD) stats -> d_out slot in (B,T,ZD) order
__global__ void kld_kernel(const float* __restrict__ pmu, const float* __restrict__ psd,
                           const float* __restrict__ emu, const float* __restrict__ esd,
                           float* __restrict__ out)
{
    int i = blockIdx.x * blockDim.x + threadIdx.x;   // T*B*ZD
    if (i >= TT_ * BB_ * ZD_) return;
    int zc = i % ZD_, rb = (i / ZD_) % BB_, tt = i / (ZD_ * BB_);
    float ps = psd[i], es = esd[i], d = emu[i] - pmu[i];
    float v = logf(ps) - logf(es) + (es * es + d * d) / (2.0f * ps * ps) - 0.5f;
    out[((size_t)rb * TT_ + tt) * ZD_ + zc] = v;
}

// =====================================================================
// host
// =====================================================================
static inline char* wsAlloc(char*& p, size_t bytes)
{ char* r = p; p += (bytes + 255) & ~(size_t)255; return r; }

extern "C" void kernel_launch(void* const* d_in, const int* in_sizes, int n_in,
                              void* d_out, int out_size, void* d_ws, size_t ws_size,
                              hipStream_t stream)
{
    (void)in_sizes; (void)n_in; (void)out_size; (void)ws_size;

    const float* x   = (const float*)d_in[0];   // (B,T,XD)
    const float* eps = (const float*)d_in[1];   // (B,T,ZD)
    // params in setup_inputs insertion order: W,b pairs
    const float* W_phi[4]  = {(const float*)d_in[2],(const float*)d_in[4],(const float*)d_in[6],(const float*)d_in[8]};
    const float* b_phi[4]  = {(const float*)d_in[3],(const float*)d_in[5],(const float*)d_in[7],(const float*)d_in[9]};
    const float* W_pr[3]   = {(const float*)d_in[10],(const float*)d_in[12],(const float*)d_in[14]};
    const float* b_pr[3]   = {(const float*)d_in[11],(const float*)d_in[13],(const float*)d_in[15]};
    const float* W_prout   = (const float*)d_in[16]; const float* b_prout = (const float*)d_in[17];
    const float* W_po[3]   = {(const float*)d_in[18],(const float*)d_in[20],(const float*)d_in[22]};
    const float* b_po[3]   = {(const float*)d_in[19],(const float*)d_in[21],(const float*)d_in[23]};
    const float* W_poout   = (const float*)d_in[24]; const float* b_poout = (const float*)d_in[25];
    const float* W_dec[3]  = {(const float*)d_in[26],(const float*)d_in[28],(const float*)d_in[30]};
    const float* b_dec[3]  = {(const float*)d_in[27],(const float*)d_in[29],(const float*)d_in[31]};
    const float* W_ih = (const float*)d_in[32]; const float* W_hh = (const float*)d_in[33];
    const float* b_ih = (const float*)d_in[34]; const float* b_hh = (const float*)d_in[35];

    char* wp = (char*)d_ws;
    // transposed bf16 weights
    __bf16* wt_phi[4]; __bf16* wt_pr[3]; __bf16* wt_po[3]; __bf16* wt_dec[3];
    wt_phi[0] = (__bf16*)wsAlloc(wp, (size_t)ZD_*HD_*2);
    for (int i=1;i<4;++i) wt_phi[i] = (__bf16*)wsAlloc(wp, (size_t)HD_*HD_*2);
    wt_pr[0] = (__bf16*)wsAlloc(wp, (size_t)RD_*HD_*2);
    for (int i=1;i<3;++i) wt_pr[i] = (__bf16*)wsAlloc(wp, (size_t)HD_*HD_*2);
    __bf16* wt_prout = (__bf16*)wsAlloc(wp, (size_t)HD_*2*ZD_*2);
    wt_po[0] = (__bf16*)wsAlloc(wp, (size_t)(RD_+XD_)*HD_*2);
    for (int i=1;i<3;++i) wt_po[i] = (__bf16*)wsAlloc(wp, (size_t)HD_*HD_*2);
    __bf16* wt_poout = (__bf16*)wsAlloc(wp, (size_t)HD_*2*ZD_*2);
    wt_dec[0] = (__bf16*)wsAlloc(wp, (size_t)(HD_+RD_)*HD_*2);
    for (int i=1;i<3;++i) wt_dec[i] = (__bf16*)wsAlloc(wp, (size_t)HD_*HD_*2);
    __bf16* wt_ih = (__bf16*)wsAlloc(wp, (size_t)(XD_+HD_)*3*RD_*2);
    __bf16* wt_hh = (__bf16*)wsAlloc(wp, (size_t)RD_*3*RD_*2);
    // activations / state
    __bf16* xbf      = (__bf16*)wsAlloc(wp, (size_t)BB_*TT_*XD_*2);
    float*  h_f32    = (float*)wsAlloc(wp, (size_t)BB_*RD_*4);
    __bf16* h_bf     = (__bf16*)wsAlloc(wp, (size_t)BB_*RD_*2);
    __bf16* h_all    = (__bf16*)wsAlloc(wp, (size_t)TT_*BB_*RD_*2);    // (T,B,RD)
    __bf16* phiz_all = (__bf16*)wsAlloc(wp, (size_t)TT_*BB_*HD_*2);    // (T,B,HD)
    float*  pmu = (float*)wsAlloc(wp, (size_t)TT_*BB_*ZD_*4);
    float*  psd = (float*)wsAlloc(wp, (size_t)TT_*BB_*ZD_*4);
    float*  emu = (float*)wsAlloc(wp, (size_t)TT_*BB_*ZD_*4);
    float*  esd = (float*)wsAlloc(wp, (size_t)TT_*BB_*ZD_*4);
    __bf16* t1  = (__bf16*)wsAlloc(wp, (size_t)BB_*HD_*2);
    __bf16* t2  = (__bf16*)wsAlloc(wp, (size_t)BB_*HD_*2);
    __bf16* t3  = (__bf16*)wsAlloc(wp, (size_t)BB_*HD_*2);
    __bf16* hx  = (__bf16*)wsAlloc(wp, (size_t)BB_*(RD_+XD_)*2);
    __bf16* xin = (__bf16*)wsAlloc(wp, (size_t)BB_*(XD_+HD_)*2);
    __bf16* zbuf= (__bf16*)wsAlloc(wp, (size_t)BB_*ZD_*2);
    float*  o_pr= (float*)wsAlloc(wp, (size_t)BB_*2*ZD_*4);
    float*  o_po= (float*)wsAlloc(wp, (size_t)BB_*2*ZD_*4);
    float*  gi  = (float*)wsAlloc(wp, (size_t)BB_*3*RD_*4);
    float*  gh  = (float*)wsAlloc(wp, (size_t)BB_*3*RD_*4);
    __bf16* dcat= (__bf16*)wsAlloc(wp, (size_t)TT_*BB_*(HD_+RD_)*2);
    __bf16* d1  = (__bf16*)wsAlloc(wp, (size_t)TT_*BB_*HD_*2);
    __bf16* d2  = (__bf16*)wsAlloc(wp, (size_t)TT_*BB_*HD_*2);

    auto xpose = [&](const float* W, __bf16* dst, int K, int N) {
        dim3 g((K + 31) / 32, (N + 31) / 32);
        hipLaunchKernelGGL(transpose_cast_bf16, g, dim3(256), 0, stream, dst, W, K, N);
    };
    // weight prep (deterministic, re-done every call)
    xpose(W_phi[0], wt_phi[0], ZD_, HD_);
    for (int i=1;i<4;++i) xpose(W_phi[i], wt_phi[i], HD_, HD_);
    xpose(W_pr[0], wt_pr[0], RD_, HD_);
    for (int i=1;i<3;++i) xpose(W_pr[i], wt_pr[i], HD_, HD_);
    xpose(W_prout, wt_prout, HD_, 2*ZD_);
    xpose(W_po[0], wt_po[0], RD_+XD_, HD_);
    for (int i=1;i<3;++i) xpose(W_po[i], wt_po[i], HD_, HD_);
    xpose(W_poout, wt_poout, HD_, 2*ZD_);
    xpose(W_dec[0], wt_dec[0], HD_+RD_, HD_);
    for (int i=1;i<3;++i) xpose(W_dec[i], wt_dec[i], HD_, HD_);
    xpose(W_ih, wt_ih, XD_+HD_, 3*RD_);
    xpose(W_hh, wt_hh, RD_, 3*RD_);

    {
        int n = BB_*TT_*XD_;
        hipLaunchKernelGGL(cast_f32_bf16, dim3((n+255)/256), dim3(256), 0, stream, xbf, x, n);
        int m = BB_*RD_;
        hipLaunchKernelGGL(zero_f32,  dim3((m+255)/256), dim3(256), 0, stream, h_f32, m);
        hipLaunchKernelGGL(zero_bf16, dim3((m+255)/256), dim3(256), 0, stream, h_bf, m);
    }

    auto gemm_rb = [&](const __bf16* A, const __bf16* WT, const float* bias,
                       __bf16* out, int M, int K, int N) {    // relu -> bf16
        hipLaunchKernelGGL((gemm_wmma<1,1>), dim3(N/256, M/64), dim3(256), 0, stream,
                           A, WT, bias, (void*)out, M, K, N, 0, 0);
    };
    auto gemm_f = [&](const __bf16* A, const __bf16* WT, const float* bias,
                      float* out, int M, int K, int N) {      // identity -> f32
        hipLaunchKernelGGL((gemm_wmma<0,0>), dim3(N/256, M/64), dim3(256), 0, stream,
                           A, WT, bias, (void*)out, M, K, N, 0, 0);
    };
    auto cpy = [&](__bf16* dst, int dld, const __bf16* src, int sld, int rows, int cols) {
        int n = rows * cols;
        hipLaunchKernelGGL(copy_bf16_blk, dim3((n+255)/256), dim3(256), 0, stream,
                           dst, dld, src, sld, rows, cols);
    };

    const int hd = BB_ * ZD_;                       // head grid elements
    for (int t = 0; t < TT_; ++t) {
        // record h_prev (bf16) for decoder
        hipMemcpyAsync(h_all + (size_t)t*BB_*RD_, h_bf, (size_t)BB_*RD_*2,
                       hipMemcpyDeviceToDevice, stream);
        // ---- prior MLP + head ----
        gemm_rb(h_bf, wt_pr[0], b_pr[0], t1, BB_, RD_, HD_);
        gemm_rb(t1,   wt_pr[1], b_pr[1], t2, BB_, HD_, HD_);
        gemm_rb(t2,   wt_pr[2], b_pr[2], t3, BB_, HD_, HD_);
        gemm_f (t3,   wt_prout, b_prout, o_pr, BB_, HD_, 2*ZD_);
        hipLaunchKernelGGL(head_prior, dim3((hd+255)/256), dim3(256), 0, stream,
                           o_pr, pmu + (size_t)t*hd, psd + (size_t)t*hd);
        // ---- posterior MLP + head (input = [h | x_t]) ----
        cpy(hx,        RD_+XD_, h_bf, RD_, BB_, RD_);
        cpy(hx + RD_,  RD_+XD_, xbf + (size_t)t*XD_, TT_*XD_, BB_, XD_);
        gemm_rb(hx, wt_po[0], b_po[0], t1, BB_, RD_+XD_, HD_);
        gemm_rb(t1, wt_po[1], b_po[1], t2, BB_, HD_, HD_);
        gemm_rb(t2, wt_po[2], b_po[2], t3, BB_, HD_, HD_);
        gemm_f (t3, wt_poout, b_poout, o_po, BB_, HD_, 2*ZD_);
        hipLaunchKernelGGL(head_post, dim3((hd+255)/256), dim3(256), 0, stream,
                           o_po, eps + (size_t)t*ZD_, TT_*ZD_,
                           emu + (size_t)t*hd, esd + (size_t)t*hd, zbuf);
        // ---- phi_z MLP (last layer writes into (T,B,HD) archive) ----
        gemm_rb(zbuf, wt_phi[0], b_phi[0], t1, BB_, ZD_, HD_);
        gemm_rb(t1,   wt_phi[1], b_phi[1], t2, BB_, HD_, HD_);
        gemm_rb(t2,   wt_phi[2], b_phi[2], t1, BB_, HD_, HD_);
        gemm_rb(t1,   wt_phi[3], b_phi[3], phiz_all + (size_t)t*BB_*HD_, BB_, HD_, HD_);
        // ---- GRU: x_in = [x_t | phi_z] ----
        cpy(xin,        XD_+HD_, xbf + (size_t)t*XD_, TT_*XD_, BB_, XD_);
        cpy(xin + XD_,  XD_+HD_, phiz_all + (size_t)t*BB_*HD_, HD_, BB_, HD_);
        gemm_f(xin,  wt_ih, b_ih, gi, BB_, XD_+HD_, 3*RD_);
        gemm_f(h_bf, wt_hh, b_hh, gh, BB_, RD_,     3*RD_);
        {
            int m = BB_ * RD_;
            hipLaunchKernelGGL(gru_combine, dim3((m+255)/256), dim3(256), 0, stream,
                               gi, gh, h_f32, h_bf);
        }
    }

    // ---- decoder over all (T,B) rows: input = [phi_z | h_prev] ----
    cpy(dcat,       HD_+RD_, phiz_all, HD_, TT_*BB_, HD_);
    cpy(dcat + HD_, HD_+RD_, h_all,    RD_, TT_*BB_, RD_);
    gemm_rb(dcat, wt_dec[0], b_dec[0], d1, TT_*BB_, HD_+RD_, HD_);
    gemm_rb(d1,   wt_dec[1], b_dec[1], d2, TT_*BB_, HD_, HD_);
    // last layer: relu, f32, rows (t*B+b) permuted to d_out (B,T,HD)
    hipLaunchKernelGGL((gemm_wmma<1,2>), dim3(HD_/256, (TT_*BB_)/64), dim3(256), 0, stream,
                       d2, wt_dec[2], b_dec[2], d_out, TT_*BB_, HD_, HD_, BB_, TT_);

    // ---- KLD into second output slot ----
    {
        int n = TT_ * BB_ * ZD_;
        float* kout = (float*)d_out + (size_t)BB_*TT_*HD_;
        hipLaunchKernelGGL(kld_kernel, dim3((n+255)/256), dim3(256), 0, stream,
                           pmu, psd, emu, esd, kout);
    }
}
